// BackwardPolicy_16947940950105
// MI455X (gfx1250) — matrix operand order repaired and checked by reading, not compiled, not run on hardware
//
#include <hip/hip_runtime.h>

#define BB   256
#define TT   1024
#define HH   512
#define PADV (-1.0f)

typedef __attribute__((ext_vector_type(16))) __bf16     v16bf;
typedef __attribute__((ext_vector_type(8)))  float      v8f;
typedef __attribute__((ext_vector_type(4)))  unsigned   u32x4;

union FragU { v16bf bf; u32x4 q[2]; };   // 32 bytes: one 16-bf16 WMMA operand per lane

__device__ __forceinline__ unsigned short f2bf(float f) {
    unsigned u = __float_as_uint(f);
    u += 0x7FFFu + ((u >> 16) & 1u);      // round-to-nearest-even
    return (unsigned short)(u >> 16);
}
__device__ __forceinline__ float bf2f(unsigned short s) {
    return __uint_as_float(((unsigned)s) << 16);
}
// Branch-free transcendentals: v_exp_f32 + v_rcp_f32, no exec-mask divergence.
__device__ __forceinline__ float sigf(float x) {
    return __builtin_amdgcn_rcpf(1.0f + __expf(-x));
}
__device__ __forceinline__ float tanhf_fast(float x) {
    // 1 - 2/(1+exp(2x)); saturates to +-1 correctly, propagates NaN, branch-free
    return 1.0f - 2.0f * __builtin_amdgcn_rcpf(1.0f + __expf(2.0f * x));
}

// ---------------------------------------------------------------------------
// Pack a row-major (nTiles*16 x 512) fp32 weight matrix (used as B^T: gates = A @ W^T)
// into bf16 B-fragments: dst[((nt*16+kc)*32+lane)*16 + j] so that each lane's
// fragment is 32 contiguous bytes. B layout (16-bit, 32x16): lanes 0-15 hold
// K=0..7,16..23 (K pairs per VGPR), lanes 16-31 hold K=8..15,24..31.
// ---------------------------------------------------------------------------
__global__ void pack_b_kernel(const float* __restrict__ src,
                              unsigned short* __restrict__ dst, int nTiles) {
    int e = blockIdx.x * blockDim.x + threadIdx.x;
    int total = nTiles * 16 * 32 * 16;
    if (e >= total) return;
    int j    = e & 15;
    int lane = (e >> 4) & 31;
    int kc   = (e >> 9) & 15;
    int nt   = e >> 13;
    int row  = nt * 16 + (lane & 15);          // output (gate) column
    int k    = kc * 32 + ((lane & 16) >> 1) + (j & 7) + ((j & 8) << 1);      // K index
    dst[e] = f2bf(src[row * HH + k]);
}

__global__ void bias_kernel(const float* __restrict__ a, const float* __restrict__ b,
                            float* __restrict__ o) {
    int i = blockIdx.x * blockDim.x + threadIdx.x;
    if (i < 4 * HH) o[i] = a[i] + b[i];
}

// ---------------------------------------------------------------------------
// LSTM recurrence. 16 workgroups, each owns 16 batch rows for all 1024 steps.
// h: double-buffered bf16 in LDS (shared A operand). c: fp32 in registers.
// 8 waves; wave w owns hidden-col tiles [4w..4w+3] and computes i,f,g,o for them.
// ---------------------------------------------------------------------------
__global__ void __launch_bounds__(256, 1)
lstm_kernel(const float* __restrict__ traj, const float* __restrict__ wih,
            const float* __restrict__ bias, const unsigned short* __restrict__ whhB,
            unsigned short* __restrict__ hOut) {
    __shared__ unsigned short hbuf[2][16][520];   // 520 stride: 16B-aligned rows, bank spread
    __shared__ float xs[16];

    const int tid   = threadIdx.x;
    const int wave  = tid >> 5, lane = tid & 31;
    const int b0    = blockIdx.x * 16;
    const int mrow  = lane & 15;
    const int rhalf = (lane & 16) ? 8 : 0;        // C/D layout: lanes>=16 are rows M+8
    const int koff  = (lane & 16) ? 8 : 0;        // A layout: lanes>=16 start at K+8
    const v8f vzero = {0.f,0.f,0.f,0.f,0.f,0.f,0.f,0.f};

    for (int i = tid; i < 2 * 16 * 520; i += 256) ((unsigned short*)hbuf)[i] = 0;

    // per-lane W_ih / (b_ih+b_hh) for the 16 owned gate-column tiles
    float wihr[16], biasr[16];
#pragma unroll
    for (int g = 0; g < 4; ++g)
#pragma unroll
        for (int hj = 0; hj < 4; ++hj) {
            int jj = g * HH + (wave * 4 + hj) * 16 + mrow;
            wihr[g * 4 + hj]  = wih[jj];
            biasr[g * 4 + hj] = bias[jj];
        }

    v8f cc[4];                                    // cell state, fp32, registers only
#pragma unroll
    for (int hj = 0; hj < 4; ++hj) cc[hj] = vzero;

    for (int t = 0; t < TT; ++t) {
        const int cur = t & 1, nxt = cur ^ 1;
        if (tid < 16) xs[tid] = traj[(b0 + tid) * TT + t];
        __syncthreads();                          // xs ready; prev h writes visible

        float xr[8];
#pragma unroll
        for (int r = 0; r < 8; ++r) xr[r] = xs[r + rhalf];

        const unsigned short* hrow = &hbuf[cur][mrow][0];

#pragma unroll
        for (int hj = 0; hj < 4; ++hj) {
            v8f acc[4];
#pragma unroll
            for (int g = 0; g < 4; ++g) acc[g] = vzero;

            for (int kc = 0; kc < 16; ++kc) {
                FragU A;
                A.q[0] = *(const u32x4*)(hrow + kc * 32 + koff);
                A.q[1] = *(const u32x4*)(hrow + kc * 32 + koff + 16);
#pragma unroll
                for (int g = 0; g < 4; ++g) {     // 4 independent WMMA chains
                    const int nt = g * 32 + wave * 4 + hj;
                    const unsigned short* bp =
                        whhB + (((size_t)(nt * 16 + kc) * 32 + lane) * 16);
                    FragU Bf;
                    Bf.q[0] = *(const u32x4*)(bp);
                    Bf.q[1] = *(const u32x4*)(bp + 8);
                    acc[g] = __builtin_amdgcn_wmma_f32_16x16x32_bf16(
                        false, A.bf, false, Bf.bf, (short)0, acc[g], false, false);
                }
            }

            const int colc = (wave * 4 + hj) * 16 + mrow;
#pragma unroll
            for (int r = 0; r < 8; ++r) {
                const int rowr = r + rhalf;
                float gi = acc[0][r] + xr[r] * wihr[0 + hj]  + biasr[0 + hj];
                float gf = acc[1][r] + xr[r] * wihr[4 + hj]  + biasr[4 + hj];
                float gg = acc[2][r] + xr[r] * wihr[8 + hj]  + biasr[8 + hj];
                float go = acc[3][r] + xr[r] * wihr[12 + hj] + biasr[12 + hj];
                float iv = sigf(gi), fv = sigf(gf);
                float gv = tanhf_fast(gg), ov = sigf(go);
                float cold = cc[hj][r];
                float cnew = fv * cold + iv * gv;
                float hnew = ov * tanhf_fast(cnew);
                bool  m    = xr[r] != PADV;       // ragged mask: freeze h,c past length
                float hold = bf2f(hbuf[cur][rowr][colc]);
                cc[hj][r] = m ? cnew : cold;
                hbuf[nxt][rowr][colc] = f2bf(m ? hnew : hold);
            }
        }
        __syncthreads();                          // h_t fully written before next step
    }

    // T=1024 even -> final h is in buffer 0
    for (int i = tid; i < 16 * 512; i += 256)
        hOut[(b0 + (i >> 9)) * 512 + (i & 511)] = hbuf[0][i >> 9][i & 511];
}

// ---------------------------------------------------------------------------
// logits = h @ fc_W.T + fc_b   (256x512 x 512x1024), WMMA bf16
// ---------------------------------------------------------------------------
__global__ void __launch_bounds__(256, 1)
fc_kernel(const unsigned short* __restrict__ hIn, const unsigned short* __restrict__ fcB,
          const float* __restrict__ fcb, float* __restrict__ logits) {
    __shared__ unsigned short hs[16][520];
    const int tid   = threadIdx.x;
    const int wave  = tid >> 5, lane = tid & 31;
    const int b0    = blockIdx.x * 16;
    const int mrow  = lane & 15;
    const int rhalf = (lane & 16) ? 8 : 0;
    const int koff  = (lane & 16) ? 8 : 0;
    const v8f vzero = {0.f,0.f,0.f,0.f,0.f,0.f,0.f,0.f};

    for (int i = tid; i < 16 * 512; i += 256)
        hs[i >> 9][i & 511] = hIn[(b0 + (i >> 9)) * 512 + (i & 511)];
    __syncthreads();

    const unsigned short* hrow = &hs[mrow][0];
    for (int it = 0; it < 8; ++it) {
        const int nt = wave * 8 + it;             // 64 col-tiles over 8 waves
        v8f acc = vzero;
        for (int kc = 0; kc < 16; ++kc) {
            FragU A, Bf;
            A.q[0] = *(const u32x4*)(hrow + kc * 32 + koff);
            A.q[1] = *(const u32x4*)(hrow + kc * 32 + koff + 16);
            const unsigned short* bp = fcB + (((size_t)(nt * 16 + kc) * 32 + lane) * 16);
            Bf.q[0] = *(const u32x4*)(bp);
            Bf.q[1] = *(const u32x4*)(bp + 8);
            acc = __builtin_amdgcn_wmma_f32_16x16x32_bf16(
                false, A.bf, false, Bf.bf, (short)0, acc, false, false);
        }
        const int col = nt * 16 + mrow;
        const float bb = fcb[col];
#pragma unroll
        for (int r = 0; r < 8; ++r)
            logits[(size_t)(b0 + r + rhalf) * TT + col] = acc[r] + bb;
    }
}

// ---------------------------------------------------------------------------
// Length-masked softmax per batch row; out = probs where t<L else 1.0
// ---------------------------------------------------------------------------
__global__ void softmax_kernel(const float* __restrict__ logits,
                               const float* __restrict__ traj,
                               float* __restrict__ out) {
    __shared__ float red[256];
    const int tid = threadIdx.x;
    const int row = blockIdx.x;
    const float* lr = logits + (size_t)row * TT;
    const float* tr = traj   + (size_t)row * TT;

    float pc = 0.f;
    for (int t = tid; t < TT; t += 256) pc += (tr[t] != PADV) ? 1.f : 0.f;
    red[tid] = pc; __syncthreads();
    for (int s = 128; s > 0; s >>= 1) { if (tid < s) red[tid] += red[tid + s]; __syncthreads(); }
    const int L = (int)red[0];
    __syncthreads();

    float pm = -3.4e38f;
    for (int t = tid; t < TT; t += 256) if (t < L) pm = fmaxf(pm, lr[t]);
    red[tid] = pm; __syncthreads();
    for (int s = 128; s > 0; s >>= 1) { if (tid < s) red[tid] = fmaxf(red[tid], red[tid + s]); __syncthreads(); }
    const float M = red[0];
    __syncthreads();

    float ps = 0.f;
    for (int t = tid; t < TT; t += 256) if (t < L) ps += __expf(lr[t] - M);
    red[tid] = ps; __syncthreads();
    for (int s = 128; s > 0; s >>= 1) { if (tid < s) red[tid] += red[tid + s]; __syncthreads(); }
    const float inv = 1.0f / red[0];

    for (int t = tid; t < TT; t += 256)
        out[(size_t)row * TT + t] = (t < L) ? __expf(lr[t] - M) * inv : 1.0f;
}

// ---------------------------------------------------------------------------
extern "C" void kernel_launch(void* const* d_in, const int* in_sizes, int n_in,
                              void* d_out, int out_size, void* d_ws, size_t ws_size,
                              hipStream_t stream) {
    const float* traj = (const float*)d_in[0];   // (256,1024) f32
    const float* W_ih = (const float*)d_in[1];   // (2048,1)
    const float* W_hh = (const float*)d_in[2];   // (2048,512)
    const float* b_ih = (const float*)d_in[3];   // (2048,)
    const float* b_hh = (const float*)d_in[4];   // (2048,)
    const float* fc_W = (const float*)d_in[5];   // (1024,512)
    const float* fc_b = (const float*)d_in[6];   // (1024,)
    float* out = (float*)d_out;

    char* ws = (char*)d_ws;
    unsigned short* whhB  = (unsigned short*)(ws);                        // 2 MB bf16 fragments
    unsigned short* fcB   = (unsigned short*)(ws + 2097152);              // 1 MB
    float*          bias  = (float*)(ws + 2097152 + 1048576);             // 8 KB
    unsigned short* hOut  = (unsigned short*)(ws + 2097152 + 1048576 + 8192);            // 256 KB
    float*          logit = (float*)(ws + 2097152 + 1048576 + 8192 + 262144);            // 1 MB

    pack_b_kernel<<<4096, 256, 0, stream>>>(W_hh, whhB, 128);
    pack_b_kernel<<<2048, 256, 0, stream>>>(fc_W, fcB, 64);
    bias_kernel<<<8, 256, 0, stream>>>(b_ih, b_hh, bias);
    lstm_kernel<<<16, 256, 0, stream>>>(traj, W_ih, bias, whhB, hOut);
    fc_kernel<<<16, 256, 0, stream>>>(hOut, fcB, fc_b, logit);
    softmax_kernel<<<256, 256, 0, stream>>>(logit, traj, out);
}
